// NT_Xent_66168266162499
// MI455X (gfx1250) — matrix operand order repaired and checked by reading, not compiled, not run on hardware
//
#include <hip/hip_runtime.h>
#include <hip/hip_bf16.h>
#include <math.h>

typedef __attribute__((ext_vector_type(16))) _Float16 v16h;
typedef __attribute__((ext_vector_type(8)))  _Float16 v8h;
typedef __attribute__((ext_vector_type(8)))  float    v8f;

#define N_TOT   8192
#define HALF_B  4096
#define DIM     256
#define CB      64              // columns staged per LDS block (32 KB per buffer)
#define NBLK    (N_TOT / CB)    // 128 column blocks total
#define CSPLIT  4               // column-range splits -> 64*4 = 256 WGs
#define BLKSPC  (NBLK / CSPLIT) // 32 blocks per split

// ---------------------------------------------------------------------------
// Kernel 1: row-normalize [z_i; z_j] into f16; zero out, Ssum[], Pos[].
// One wave32 per row; 8 f32 per lane (32*8 = 256 = DIM).
// ---------------------------------------------------------------------------
__global__ __launch_bounds__(256) void nt_xent_normalize(
    const float* __restrict__ zi, const float* __restrict__ zj,
    _Float16* __restrict__ zn, float* __restrict__ Ssum,
    float* __restrict__ Pos, float* __restrict__ out)
{
    const int w    = threadIdx.x >> 5;
    const int lane = threadIdx.x & 31;
    const int row  = blockIdx.x * 8 + w;

    // zero the per-row accumulators (blocks 0..31 cover 8192 rows)
    if (blockIdx.x < 32) {
        const int i = blockIdx.x * 256 + threadIdx.x;
        Ssum[i] = 0.0f;
        Pos[i]  = 0.0f;
    }
    if (blockIdx.x == 0 && threadIdx.x == 0) *out = 0.0f;

    const float* src = (row < HALF_B) ? (zi + (size_t)row * DIM)
                                      : (zj + (size_t)(row - HALF_B) * DIM);
    const float4 a = *(const float4*)(src + lane * 8);
    const float4 b = *(const float4*)(src + lane * 8 + 4);

    float ss = a.x*a.x + a.y*a.y + a.z*a.z + a.w*a.w
             + b.x*b.x + b.y*b.y + b.z*b.z + b.w*b.w;
    #pragma unroll
    for (int m = 16; m >= 1; m >>= 1) ss += __shfl_xor(ss, m, 32);

    const float inv = 1.0f / fmaxf(sqrtf(ss), 1e-8f);

    v8h o;
    o[0] = (_Float16)(a.x * inv); o[1] = (_Float16)(a.y * inv);
    o[2] = (_Float16)(a.z * inv); o[3] = (_Float16)(a.w * inv);
    o[4] = (_Float16)(b.x * inv); o[5] = (_Float16)(b.y * inv);
    o[6] = (_Float16)(b.z * inv); o[7] = (_Float16)(b.w * inv);
    *(v8h*)(zn + (size_t)row * DIM + lane * 8) = o;
}

// ---------------------------------------------------------------------------
// Async-stage one 64-column (= 64-row of zn) block, 32 KB, into LDS.
// 256 threads x 8 chunks x 16 B, CDNA5 ASYNCcnt path (no VGPR data staging).
// XOR-swizzle each 16 B chunk by (row&15)<<4 so later per-row ds_load_b128
// across 16 lanes lands on all 64 banks conflict-free (no padding needed).
// ---------------------------------------------------------------------------
__device__ __forceinline__ void stage_block(const _Float16* __restrict__ zn,
                                            unsigned char* smbase,
                                            int tid, int cbk)
{
    // ISA flat-aperture rule: LDS offset = generic address truncated to 32 bits.
    const uint32_t lbase  = (uint32_t)(uintptr_t)smbase;
    const uint32_t gblock = (uint32_t)cbk * (uint32_t)(CB * DIM * 2);
    #pragma unroll
    for (int i = 0; i < 8; ++i) {
        const int      c     = tid + i * 256;          // 16B-chunk index in block
        const uint32_t row   = (uint32_t)(c >> 5);     // 32 chunks per 512B row
        const uint32_t o     = (uint32_t)(c & 31) << 4;
        const uint32_t laddr = lbase + row * 512u + (o ^ ((row & 15u) << 4));
        const uint32_t gaddr = gblock + ((uint32_t)c << 4);
        asm volatile("global_load_async_to_lds_b128 %0, %1, %2"
                     :: "v"(laddr), "v"(gaddr), "s"(zn)
                     : "memory");
    }
}

// ---------------------------------------------------------------------------
// Kernel 2: fused sim GEMM (WMMA f16 -> f32) + bounded-shift partial softmax
// sums + positive-pair extraction. sim/T = 2*cos in [-2,2], so per-row
// Sum exp(v-2) needs no running max (exp arg in [-4,0], sum <= 8191:
// f32-safe) and partial column-range sums combine by plain addition --
// letting the GEMM split over columns for full-chip occupancy (256 WGs).
//
// Grid: 64 row-blocks x 4 column-splits. 8 waves/WG, wave owns a 16-row
// strip (WG = 128 rows). A panel (16x256 f16) resident in VGPRs; B
// double-buffered in LDS via async loads; 32 WMMAs/wave per stage.
// ---------------------------------------------------------------------------
__global__ __launch_bounds__(256) void nt_xent_wmma(
    const _Float16* __restrict__ zn, float* __restrict__ Ssum,
    float* __restrict__ Pos)
{
    __shared__ __align__(16) unsigned char smem[2][CB * DIM * 2];  // 2 x 32 KB

    const int tid  = threadIdx.x;
    const int w    = tid >> 5;
    const int lane = tid & 31;
    const int half = lane >> 4;      // k-group selector per ISA VGPR layout
    const int l16  = lane & 15;
    const int rowbase = blockIdx.x * 128 + w * 16;
    const int cbk0    = blockIdx.y * BLKSPC;

    // ---- A fragments: rows rowbase..rowbase+15, K = 0..255 (ISA 16-bit A
    // 16x32 layout: lanes 0-15 hold K 0-7 & 16-23, lanes 16-31 K 8-15 & 24-31)
    const _Float16* aptr = zn + (size_t)(rowbase + l16) * DIM;
    v16h afrag[8];
    #pragma unroll
    for (int ks = 0; ks < 8; ++ks) {
        v8h lo = *(const v8h*)(aptr + ks * 32 + half * 8);
        v8h hi = *(const v8h*)(aptr + ks * 32 + 16 + half * 8);
        afrag[ks] = __builtin_shufflevector(lo, hi,
            0,1,2,3,4,5,6,7,8,9,10,11,12,13,14,15);
    }

    float runsum[8], posv[8];
    #pragma unroll
    for (int r = 0; r < 8; ++r) { runsum[r] = 0.0f; posv[r] = 0.0f; }

    stage_block(zn, &smem[0][0], tid, cbk0);

    #pragma unroll 1
    for (int it = 0; it < BLKSPC; ++it) {
        const int cbk = cbk0 + it;
        const int buf = it & 1;
        asm volatile("s_wait_asynccnt 0" ::: "memory");  // my stage of `buf` done
        __syncthreads();           // all waves staged `buf`; prev compute done
        if (it + 1 < BLKSPC) stage_block(zn, &smem[buf ^ 1][0], tid, cbk + 1);

        const unsigned char* bb = &smem[buf][0];
        const uint32_t sw = (uint32_t)l16 << 4;   // read-side un-swizzle

        #pragma unroll
        for (int t = 0; t < 4; ++t) {
            // B 32x16 layout: lane = column (row t*16+l16 of the block),
            // k-group = half: lane holds K = half*8 + 0..7 & 16 + half*8 + 0..7 per 32-wide step.
            const unsigned char* rp = bb + (t * 16 + l16) * 512;

            v8f acc = {0.f,0.f,0.f,0.f,0.f,0.f,0.f,0.f};
            #pragma unroll
            for (int ks = 0; ks < 8; ++ks) {
                const uint32_t o = (uint32_t)(half * 16 + ks * 64);
                v8h lo = *(const v8h*)(rp + (o ^ sw));
                v8h hi = *(const v8h*)(rp + ((o + 32u) ^ sw));
                v16h bfrag = __builtin_shufflevector(lo, hi,
                    0,1,2,3,4,5,6,7,8,9,10,11,12,13,14,15);
                acc = __builtin_amdgcn_wmma_f32_16x16x32_f16(
                    false, afrag[ks], false, bfrag, (short)0, acc, false, false);
            }

            // C/D layout: VGPR r holds M = half*8 + r, N = l16.
            const int n = cbk * CB + t * 16 + l16;
            #pragma unroll
            for (int r = 0; r < 8; ++r) {
                const int   m = rowbase + half * 8 + r;
                const float v = acc[r] * 2.0f;           // 1/T
                const float e = __expf(v - 2.0f);        // bounded shift
                if (n != m) runsum[r] += e;              // exact self-mask
                if (n == ((m + HALF_B) & (N_TOT - 1))) posv[r] += v;
            }
        }
    }

    // ---- merge the 16 lanes of each half-wave (they share rows), then
    // accumulate this WG's column-range partials into the per-row arrays ----
    #pragma unroll
    for (int r = 0; r < 8; ++r) {
        float s0 = runsum[r], p0 = posv[r];
        #pragma unroll
        for (int msk = 1; msk < 16; msk <<= 1) {
            s0 += __shfl_xor(s0, msk, 32);
            p0 += __shfl_xor(p0, msk, 32);
        }
        if (l16 == 0) {
            const int m = rowbase + half * 8 + r;
            atomicAdd(Ssum + m, s0);
            atomicAdd(Pos + m, p0);
        }
    }
}

// ---------------------------------------------------------------------------
// Kernel 3: loss = sum_i (2 + log(Ssum[i]) - Pos[i]) / N
// ---------------------------------------------------------------------------
__global__ __launch_bounds__(256) void nt_xent_finish(
    const float* __restrict__ Ssum, const float* __restrict__ Pos,
    float* __restrict__ out)
{
    const int i = blockIdx.x * 256 + threadIdx.x;
    float v = (2.0f + __logf(Ssum[i])) - Pos[i];
    #pragma unroll
    for (int m = 16; m >= 1; m >>= 1) v += __shfl_xor(v, m, 32);
    if ((threadIdx.x & 31) == 0) {
        atomicAdd(out, v * (1.0f / (float)N_TOT));
    }
}

// ---------------------------------------------------------------------------
extern "C" void kernel_launch(void* const* d_in, const int* in_sizes, int n_in,
                              void* d_out, int out_size, void* d_ws, size_t ws_size,
                              hipStream_t stream)
{
    const float* zi = (const float*)d_in[0];
    const float* zj = (const float*)d_in[1];
    float* out      = (float*)d_out;

    _Float16* zn = (_Float16*)d_ws;                       // 4 MB f16 normalized
    float* Ssum  = (float*)((char*)d_ws + (size_t)N_TOT * DIM * 2);  // 32 KB
    float* Pos   = Ssum + N_TOT;                                     // 32 KB

    nt_xent_normalize<<<dim3(N_TOT / 8), dim3(256), 0, stream>>>(zi, zj, zn, Ssum, Pos, out);
    nt_xent_wmma<<<dim3(N_TOT / 128, CSPLIT), dim3(256), 0, stream>>>(zn, Ssum, Pos);
    nt_xent_finish<<<dim3(N_TOT / 256), dim3(256), 0, stream>>>(Ssum, Pos, out);
}
